// TreeClassifier_30296699306647
// MI455X (gfx1250) — matrix operand order repaired and checked by reading, compiled, not run on hardware
//
#include <hip/hip_runtime.h>
#include <hip/hip_bf16.h>

// ---------------------------------------------------------------------------
// Soft decision-tree forest classifier, MI455X (gfx1250, wave32, WMMA bf16)
//
//   GEMM1 (fused): x[8192,1024] @ sw^T -> sigmoid -> 4-deep leaf products
//                  -> lp[8192,512] bf16.  Wave tile 64Mx16N (1 tree),
//                  double-buffered LDS A staging, 1 barrier / k-step.
//   GEMM2:         lp[8192,512] @ WLDt^T[512,112] -> out[8192,100] f32.
//                  Whole 112KB B matrix staged in LDS via async-to-LDS.
// ---------------------------------------------------------------------------

typedef __attribute__((ext_vector_type(16))) __bf16 v16bf;
typedef __attribute__((ext_vector_type(8)))  float  v8f;

#define BATCH   8192
#define DFEAT   1024
#define NTREES  32
#define NINT    15
#define NLEAF   16
#define NCLS    100
#define NCLSP   112                   // 100 padded to 7 WMMA n-tiles
#define K2      512                   // NTREES * NLEAF
#define WLDT_BYTES (NCLSP * K2 * 2)   // 114688

#define A_STRIDE 40                   // 64x32 bf16 tile rows padded: 80B = 20 banks
#define P_STRIDE 17                   // sigmoid tile rows padded: 68B = 17 banks
#define B2_STRIDE 520                 // WLDt rows in LDS: 1040B = 4-bank rotation
#define LSB_BYTES (NCLSP * B2_STRIDE * 2)   // 116480

#if __has_builtin(__builtin_amdgcn_global_load_async_to_lds_b128) && \
    __has_builtin(__builtin_amdgcn_s_wait_asynccnt)
#define USE_ASYNC_LDS 1
#else
#define USE_ASYNC_LDS 0
#endif

#if USE_ASYNC_LDS
// Builtin signature (from hipcc diagnostic): ptr args are int4-vector pointers
// in explicit address spaces: (v4i AS(1)*, v4i AS(3)*, imm offset, imm cpol).
typedef int v4i __attribute__((ext_vector_type(4)));
typedef __attribute__((address_space(1))) v4i* as1_v4i_p;
typedef __attribute__((address_space(3))) v4i* as3_v4i_p;

__device__ __forceinline__ void async_copy16(const void* gsrc, void* ldst)
{
    __builtin_amdgcn_global_load_async_to_lds_b128(
        (as1_v4i_p)gsrc, (as3_v4i_p)ldst, 0, 0);
}
#endif

__device__ __forceinline__ float fast_rcp(float a)
{
#if __has_builtin(__builtin_amdgcn_rcpf)
    return __builtin_amdgcn_rcpf(a);      // v_rcp_f32: 1 TRANS op, ~1 ulp
#else
    return 1.0f / a;
#endif
}

__device__ __forceinline__ float fast_sigmoid(float z)
{
    return fast_rcp(1.0f + __expf(-z));
}

// ------------------------------ prep ---------------------------------------
// One block of 512 threads; thread = t*16 + l. Produces transposed, padded,
// tree-weighted leaf distribution WLDt[c][t*16+l] in bf16.
__global__ __launch_bounds__(512)
void TreePrep_kernel(const float* __restrict__ ll, const float* __restrict__ tw,
                     const float* __restrict__ log_temp, __bf16* __restrict__ wldt)
{
    const int tid = threadIdx.x;          // 0..511
    const int t   = tid >> 4;

    float temp = __expf(log_temp[0]);
    temp = fminf(fmaxf(temp, 0.1f), 5.0f);
    const float invt = 1.0f / temp;

    float twmax = -3.4e38f;
    for (int j = 0; j < NTREES; ++j) twmax = fmaxf(twmax, tw[j]);
    float tsum = 0.f;
    for (int j = 0; j < NTREES; ++j) tsum += __expf(tw[j] - twmax);
    const float w = __expf(tw[t] - twmax) / tsum;

    const float* llp = ll + (size_t)tid * NCLS;
    float mx = -3.4e38f;
    for (int c = 0; c < NCLS; ++c) mx = fmaxf(mx, llp[c]);
    float se = 0.f;
    for (int c = 0; c < NCLS; ++c) se += __expf((llp[c] - mx) * invt);
    const float scale = w / se;

    for (int c = 0; c < NCLSP; ++c) {
        const float v = (c < NCLS) ? scale * __expf((llp[c] - mx) * invt) : 0.f;
        wldt[(size_t)c * K2 + tid] = (__bf16)v;
    }
}

// ------------------------------ GEMM1 (fused) -------------------------------
// grid = (BATCH/64, NTREES/8), block = 256 (8 waves).
// Wave w: rows [mb, mb+64) x tree (by*8 + w); 4 WMMA accumulators.
__global__ __launch_bounds__(256)
void TreeGemm1_kernel(const float* __restrict__ x,  const float* __restrict__ sw,
                      const float* __restrict__ sb, const float* __restrict__ log_temp,
                      __bf16* __restrict__ lp)
{
    __shared__ __bf16 lsA[2][64 * A_STRIDE];    // double-buffered x tile (bf16)
    __shared__ float  lsP[8][16 * P_STRIDE];    // per-wave sigmoid tile

    const int tid  = threadIdx.x;
    const int wave = tid >> 5;
    const int lane = tid & 31;
    const int r    = lane & 15;
    const int h    = lane >> 4;
    const int mb   = blockIdx.x * 64;
    const int tree = blockIdx.y * 8 + wave;

    float temp = __expf(log_temp[0]);
    temp = fminf(fmaxf(temp, 0.1f), 5.0f);
    const float invtemp = 1.0f / temp;

    // Lane 15 duplicates a valid row: C column 15 is never consumed below,
    // so no zeroing / EXEC divergence is needed.
    const int    br    = (r < NINT) ? r : 0;
    const float* browp = sw + (size_t)(tree * NINT + br) * DFEAT;

    v8f acc[4];
    #pragma unroll
    for (int st = 0; st < 4; ++st)
        acc[st] = (v8f){0.f,0.f,0.f,0.f,0.f,0.f,0.f,0.f};

    // A staging map: 256 threads x 8 f32 = 64x32 tile per buffer
    const int    srow = tid >> 2;
    const int    skk  = (tid & 3) * 8;
    const float* xsp  = x + (size_t)(mb + srow) * DFEAT + skk;

    // prologue: stage k-chunk 0 into buffer 0
    {
        const float4 g0 = *(const float4*)(xsp);
        const float4 g1 = *(const float4*)(xsp + 4);
        union { __bf16 b[8]; uint4 q; } pk;
        pk.b[0]=(__bf16)g0.x; pk.b[1]=(__bf16)g0.y; pk.b[2]=(__bf16)g0.z; pk.b[3]=(__bf16)g0.w;
        pk.b[4]=(__bf16)g1.x; pk.b[5]=(__bf16)g1.y; pk.b[6]=(__bf16)g1.z; pk.b[7]=(__bf16)g1.w;
        *(uint4*)&lsA[0][srow * A_STRIDE + skk] = pk.q;
    }

    for (int kk = 0; kk < DFEAT / 32; ++kk) {
        const int k0  = kk * 32;
        const int cur = kk & 1;
        __syncthreads();      // buf[cur] staged; buf[cur^1] readers (kk-1) done

        // issue next tile's global loads early (latency hides under WMMAs);
        // last iteration re-loads the current chunk (staged result unused)
        const int    knext = (k0 + 32 < DFEAT) ? (k0 + 32) : k0;
        const float4 g0 = *(const float4*)(xsp + knext);
        const float4 g1 = *(const float4*)(xsp + knext + 4);

        // B fragment: lane (r,h) holds sw row (tree*15+br), K = k0+16h .. +15
        v16bf bv;
        const float* bp = browp + k0 + 16 * h;
        #pragma unroll
        for (int j = 0; j < 16; j += 4) {
            const float4 f = *(const float4*)(bp + j);
            bv[j]     = (__bf16)f.x; bv[j + 1] = (__bf16)f.y;
            bv[j + 2] = (__bf16)f.z; bv[j + 3] = (__bf16)f.w;
        }
        __builtin_prefetch(bp + 32, 0, 3);

        // A fragments from LDS + 4 WMMAs sharing one B fragment
        #pragma unroll
        for (int st = 0; st < 4; ++st) {
            union { v16bf v; uint4 q[2]; } ua;
            const __bf16* arow = &lsA[cur][(st * 16 + r) * A_STRIDE];
            ua.q[0] = *(const uint4*)(arow + 16 * h);
            ua.q[1] = *(const uint4*)(arow + 16 * h + 8);
            acc[st] = __builtin_amdgcn_wmma_f32_16x16x32_bf16(
                          false, ua.v, false, bv, (short)0, acc[st], false, false);
        }

        // stage next chunk into the other buffer (barrier at next top orders it)
        {
            union { __bf16 b[8]; uint4 q; } pk;
            pk.b[0]=(__bf16)g0.x; pk.b[1]=(__bf16)g0.y; pk.b[2]=(__bf16)g0.z; pk.b[3]=(__bf16)g0.w;
            pk.b[4]=(__bf16)g1.x; pk.b[5]=(__bf16)g1.y; pk.b[6]=(__bf16)g1.z; pk.b[7]=(__bf16)g1.w;
            *(uint4*)&lsA[cur ^ 1][srow * A_STRIDE + skk] = pk.q;
        }
    }

    // ---- epilogue: bias + sigmoid, wave-local transpose, leaf products ----
    const float sbv = sb[tree * NINT + br];
    float* spw = &lsP[wave][0];
    #pragma unroll
    for (int st = 0; st < 4; ++st) {
        __syncthreads();
        #pragma unroll
        for (int c = 0; c < 8; ++c) {             // C layout: row c+8h, col r
            const float z = (acc[st][c] + sbv) * invtemp;
            spw[(c + 8 * h) * P_STRIDE + r] = fast_sigmoid(z);
        }
        __syncthreads();

        float spv[NINT];
        #pragma unroll
        for (int i = 0; i < NINT; ++i) spv[i] = spw[r * P_STRIDE + i];

        union { __bf16 b[8]; uint4 q; } lf;       // leaves l = 8h .. 8h+7
        #pragma unroll
        for (int j = 0; j < 8; ++j) {
            const int l = 8 * h + j;
            float p = 1.f;
            int  ni = l + NINT;
            #pragma unroll
            for (int d = 0; d < 4; ++d) {
                const int pi = (ni - 1) >> 1;
                p *= (ni & 1) ? (1.f - spv[pi]) : spv[pi];
                ni = pi;
            }
            lf.b[j] = (__bf16)p;
        }
        *(uint4*)(lp + (size_t)(mb + st * 16 + r) * K2 + tree * NLEAF + 8 * h) = lf.q;
    }
}

// ------------------------------ GEMM2 ---------------------------------------
// grid = 64, block = 256 (8 waves). Wave tile: 16 rows x 112 cols, K=512.
// Entire WLDt (112KB) staged in LDS (async-to-LDS when available), rows padded
// to 520 bf16 so 16-lane column reads rotate 4 banks/row (conflict-free).
__global__ __launch_bounds__(256)
void TreeGemm2_kernel(const __bf16* __restrict__ lp, const __bf16* __restrict__ wldt,
                      float* __restrict__ out)
{
    extern __shared__ __bf16 lsB[];               // NCLSP * B2_STRIDE elements
    const int tid  = threadIdx.x;
    const int wave = tid >> 5;
    const int lane = tid & 31;
    const int r    = lane & 15;
    const int h    = lane >> 4;
    const int m0   = (blockIdx.x * 8 + wave) * 16;

    // stage WLDt: 7168 16B-chunks over 256 threads = 28 iterations
    #pragma unroll 4
    for (int it = 0; it < (NCLSP * K2) / (256 * 8); ++it) {
        const int cid = it * 256 + tid;
        const int row = cid >> 6;
        const int col = (cid & 63) * 8;
        const __bf16* gp = wldt + (size_t)row * K2 + col;
        __bf16*       sp = lsB + (size_t)row * B2_STRIDE + col;
#if USE_ASYNC_LDS
        async_copy16(gp, sp);
#else
        *(uint4*)sp = *(const uint4*)gp;
#endif
    }
#if USE_ASYNC_LDS
    __builtin_amdgcn_s_wait_asynccnt(0);
#endif
    __syncthreads();

    v8f acc[7];
    #pragma unroll
    for (int nt = 0; nt < 7; ++nt)
        acc[nt] = (v8f){0.f,0.f,0.f,0.f,0.f,0.f,0.f,0.f};

    for (int k0 = 0; k0 < K2; k0 += 32) {
        const __bf16* ap = lp + (size_t)(m0 + r) * K2 + k0;
        union { v16bf v; uint4 q[2]; } ua;
        ua.q[0] = *(const uint4*)(ap + 16 * h);
        ua.q[1] = *(const uint4*)(ap + 16 * h + 8);

        #pragma unroll
        for (int nt = 0; nt < 7; ++nt) {
            const __bf16* bp = &lsB[(nt * 16 + r) * B2_STRIDE + k0 + 16 * h];
            union { v16bf v; uint4 q[2]; } ub;
            ub.q[0] = *(const uint4*)bp;
            ub.q[1] = *(const uint4*)(bp + 8);
            acc[nt] = __builtin_amdgcn_wmma_f32_16x16x32_bf16(
                          false, ua.v, false, ub.v, (short)0, acc[nt], false, false);
        }
    }

    #pragma unroll
    for (int nt = 0; nt < 7; ++nt) {
        const int c = nt * 16 + r;
        if (c < NCLS) {
            #pragma unroll
            for (int cc = 0; cc < 8; ++cc)
                out[(size_t)(m0 + cc + 8 * h) * NCLS + c] = acc[nt][cc];
        }
    }
}

// ------------------------------ launcher ------------------------------------
extern "C" void kernel_launch(void* const* d_in, const int* in_sizes, int n_in,
                              void* d_out, int out_size, void* d_ws, size_t ws_size,
                              hipStream_t stream)
{
    const float* x  = (const float*)d_in[0];
    const float* sw = (const float*)d_in[1];
    const float* sb = (const float*)d_in[2];
    const float* ll = (const float*)d_in[3];
    const float* tw = (const float*)d_in[4];
    const float* lt = (const float*)d_in[5];
    float* out = (float*)d_out;

    __bf16* wldt = (__bf16*)d_ws;
    __bf16* lp   = (__bf16*)((char*)d_ws + WLDT_BYTES);

    TreePrep_kernel<<<1, 512, 0, stream>>>(ll, tw, lt, wldt);

    dim3 g1(BATCH / 64, NTREES / 8);                 // 128 x 4 blocks
    TreeGemm1_kernel<<<g1, 256, 0, stream>>>(x, sw, sb, lt, lp);

    TreeGemm2_kernel<<<BATCH / 128, 256, LSB_BYTES, stream>>>(lp, wldt, out);
}
